// AttentiveTransformer_58033598103868
// MI455X (gfx1250) — matrix operand order repaired and checked by reading, not compiled, not run on hardware
//
#include <hip/hip_runtime.h>
#include <hip/hip_bf16.h>

typedef __attribute__((ext_vector_type(16))) __bf16 v16bf;
typedef __attribute__((ext_vector_type(8)))  float  v8f;

#define B_TOTAL 65536
#define IN_DIM  512
#define G_DIM   256
#define VBS     128
#define KSTEP   32
#define BN_EPS  1e-5f

// Dynamic LDS layout:
//   [0, 16384)            : W chunk staged as bf16  [256 g][32 k]   (GEMM phase)
//   [0, 131072)           : x tile f32 [128 m][256 n]               (epilogue phase, aliases W stage)
//   [131072, 133120)      : BN stats: sum[256], sumsq[256]
#define SMEM_BYTES (131072 + 2048)

__global__ __launch_bounds__(256) void attentive_transformer_kernel(
    const float* __restrict__ priors,   // [B, G]
    const float* __restrict__ feat,     // [B, IN]
    const float* __restrict__ W,        // [G, IN]
    const float* __restrict__ gamma,    // [G]
    const float* __restrict__ beta,     // [G]
    float* __restrict__ out)            // [B, G]
{
    extern __shared__ char smem_raw[];
    __bf16* lds_w    = (__bf16*)smem_raw;               // GEMM phase
    float*  xtile    = (float*)smem_raw;                // epilogue phase (alias)
    float*  stat_sum = (float*)(smem_raw + 131072);
    float*  stat_sq  = stat_sum + G_DIM;

    const int tid  = threadIdx.x;
    const int lane = tid & 31;
    const int wave = tid >> 5;           // 0..7
    const int mrow = lane & 15;          // A: row-in-16; B/C: col-in-16
    const int hi   = lane >> 4;          // lane half (K/row-offset selector)
    const int row0 = blockIdx.x * VBS;   // virtual batch start row

    // --- GEMM: 16 accumulator tiles of 16x16 per wave (rows wave*16..wave*16+15, all 256 cols)
    v8f acc[16] = {};

    const float* arow = feat + (size_t)(row0 + wave * 16 + mrow) * IN_DIM;

    for (int kc = 0; kc < IN_DIM; kc += KSTEP) {
        // Stage W[:, kc..kc+31] into LDS as bf16. thread g==tid handles one W row.
        __syncthreads();
        {
            const float* wrow = W + (size_t)tid * IN_DIM + kc;
            __bf16* d = lds_w + tid * KSTEP;
            #pragma unroll
            for (int i = 0; i < 8; ++i) {
                float4 w4 = *(const float4*)(wrow + 4 * i);
                d[4 * i + 0] = (__bf16)w4.x;
                d[4 * i + 1] = (__bf16)w4.y;
                d[4 * i + 2] = (__bf16)w4.z;
                d[4 * i + 3] = (__bf16)w4.w;
            }
        }
        __syncthreads();

        // A fragment in contiguous K order (matches B):
        // lanes 0-15: elems 0..15 = K kc..kc+15
        // lanes 16-31: elems 0..15 = K kc+16..kc+31
        const float* ap = arow + kc + hi * 16;
        float4 a0 = *(const float4*)(ap + 0);
        float4 a1 = *(const float4*)(ap + 4);
        float4 a2 = *(const float4*)(ap + 8);
        float4 a3 = *(const float4*)(ap + 12);
        if (kc + KSTEP < IN_DIM) __builtin_prefetch(ap + KSTEP, 0, 3);

        v16bf afrag;
        afrag[0]  = (__bf16)a0.x; afrag[1]  = (__bf16)a0.y;
        afrag[2]  = (__bf16)a0.z; afrag[3]  = (__bf16)a0.w;
        afrag[4]  = (__bf16)a1.x; afrag[5]  = (__bf16)a1.y;
        afrag[6]  = (__bf16)a1.z; afrag[7]  = (__bf16)a1.w;
        afrag[8]  = (__bf16)a2.x; afrag[9]  = (__bf16)a2.y;
        afrag[10] = (__bf16)a2.z; afrag[11] = (__bf16)a2.w;
        afrag[12] = (__bf16)a3.x; afrag[13] = (__bf16)a3.y;
        afrag[14] = (__bf16)a3.z; afrag[15] = (__bf16)a3.w;

        // 16 column tiles: B fragment = 16 contiguous bf16 (K) for column g = t*16+mrow,
        // lanes 0-15 cover K 0..15, lanes 16-31 cover K 16..31.
        #pragma unroll
        for (int t = 0; t < 16; ++t) {
            const v16bf bfrag =
                *(const v16bf*)(lds_w + (t * 16 + mrow) * KSTEP + hi * 16);
            acc[t] = __builtin_amdgcn_wmma_f32_16x16x32_bf16(
                false, afrag, false, bfrag, (short)0, acc[t], false, false);
        }
    }

    // --- Ghost BN stats: per-column sum / sumsq over the 128 rows of this virtual batch
    stat_sum[tid] = 0.0f;
    stat_sq[tid]  = 0.0f;
    __syncthreads();   // also: all waves done reading lds_w -> safe to alias as xtile

    #pragma unroll
    for (int t = 0; t < 16; ++t) {
        float s = 0.0f, q = 0.0f;
        #pragma unroll
        for (int p = 0; p < 8; ++p) {
            float v = acc[t][p];
            s += v;
            q += v * v;
        }
        // combine lane halves (rows 0-7 with rows 8-15 of this wave's 16 rows)
        s += __shfl_xor(s, 16, 32);
        q += __shfl_xor(q, 16, 32);
        if (hi == 0) {
            atomicAdd(&stat_sum[t * 16 + mrow], s);   // ds_add_f32
            atomicAdd(&stat_sq[t * 16 + mrow], q);
        }
    }
    __syncthreads();

    // --- Normalize, gamma/beta, priors; scatter to LDS x tile (row-major for sparsemax)
    const float inv_vbs = 1.0f / (float)VBS;
    #pragma unroll
    for (int t = 0; t < 16; ++t) {
        const int   n    = t * 16 + mrow;
        const float mean = stat_sum[n] * inv_vbs;
        const float var  = stat_sq[n] * inv_vbs - mean * mean;
        const float rstd = __frsqrt_rn(var + BN_EPS);
        const float gm   = gamma[n];
        const float bt   = beta[n];
        #pragma unroll
        for (int p = 0; p < 8; ++p) {
            const int m = wave * 16 + hi * 8 + p;     // row within tile (C-layout)
            float v = (acc[t][p] - mean) * rstd * gm + bt;
            v *= priors[(size_t)(row0 + m) * G_DIM + n];
            xtile[m * G_DIM + n] = v;
        }
    }
    __syncthreads();

    // --- Sparsemax per row: 8 lanes per row (32 values each, in registers),
    // tau via bisection of sum(relu(z - tau)) = 1 on [max-1, max].
    const int grp = lane >> 3;   // 0..3 : which of 4 concurrent rows
    const int sub = lane & 7;    // 0..7 : lane within row group
    for (int j = 0; j < 4; ++j) {
        const int m = wave * 16 + j * 4 + grp;
        const float* xr = xtile + m * G_DIM + sub * 32;

        float z[32];
        #pragma unroll
        for (int i = 0; i < 8; ++i) {
            float4 v = *(const float4*)(xr + 4 * i);
            z[4 * i + 0] = v.x; z[4 * i + 1] = v.y;
            z[4 * i + 2] = v.z; z[4 * i + 3] = v.w;
        }

        float mx = z[0];
        #pragma unroll
        for (int i = 1; i < 32; ++i) mx = fmaxf(mx, z[i]);
        mx = fmaxf(mx, __shfl_xor(mx, 1, 32));
        mx = fmaxf(mx, __shfl_xor(mx, 2, 32));
        mx = fmaxf(mx, __shfl_xor(mx, 4, 32));

        float lo = mx - 1.0f, hiT = mx;
        #pragma unroll 1
        for (int it = 0; it < 26; ++it) {
            const float mid = 0.5f * (lo + hiT);
            float s = 0.0f;
            #pragma unroll
            for (int i = 0; i < 32; ++i) s += fmaxf(z[i] - mid, 0.0f);
            s += __shfl_xor(s, 1, 32);
            s += __shfl_xor(s, 2, 32);
            s += __shfl_xor(s, 4, 32);
            if (s >= 1.0f) lo = mid; else hiT = mid;
        }
        const float tau = lo;

        float* orow = out + (size_t)(row0 + m) * G_DIM + sub * 32;
        #pragma unroll
        for (int i = 0; i < 8; ++i) {
            float4 v;
            v.x = fmaxf(z[4 * i + 0] - tau, 0.0f);
            v.y = fmaxf(z[4 * i + 1] - tau, 0.0f);
            v.z = fmaxf(z[4 * i + 2] - tau, 0.0f);
            v.w = fmaxf(z[4 * i + 3] - tau, 0.0f);
            *(float4*)(orow + 4 * i) = v;
        }
    }
}

extern "C" void kernel_launch(void* const* d_in, const int* in_sizes, int n_in,
                              void* d_out, int out_size, void* d_ws, size_t ws_size,
                              hipStream_t stream) {
    (void)in_sizes; (void)n_in; (void)d_ws; (void)ws_size; (void)out_size;
    const float* priors = (const float*)d_in[0];
    const float* feat   = (const float*)d_in[1];
    const float* W      = (const float*)d_in[2];
    const float* gamma  = (const float*)d_in[3];
    const float* beta   = (const float*)d_in[4];
    float* out = (float*)d_out;

    dim3 grid(B_TOTAL / VBS);   // 512 workgroups, one per ghost-BN virtual batch
    dim3 block(256);            // 8 wave32 waves
    attentive_transformer_kernel<<<grid, block, SMEM_BYTES, stream>>>(
        priors, feat, W, gamma, beta, out);
}
